// LLaDA_TransformerBlock_60576218743084
// MI455X (gfx1250) — matrix operand order repaired and checked by reading, not compiled, not run on hardware
//
#include <hip/hip_runtime.h>
#include <cstdint>
#include <cstddef>

// Problem constants (LLaDA block): B=2, S=2048, D=2048, H=16, HD=128, I=4096
#define B_  2
#define S_  2048
#define D_  2048
#define H_  16
#define HD_ 128
#define I_  4096
#define NT_ (B_ * S_)      // 4096 tokens
#define EPS_ 1e-5f

typedef unsigned short u16;
typedef unsigned int   u32;
typedef unsigned long long u64;
typedef __attribute__((ext_vector_type(4)))  u32    u32x4;
typedef __attribute__((ext_vector_type(8)))  float  v8f;
typedef __attribute__((ext_vector_type(16))) __bf16 v16bf;

union FragAB { v16bf v; u32x4 q[2]; u16 h[16]; };
union FragC  { v8f   v; float f[8]; };
union U4H8   { u32x4 q; u16 h[8]; };

__device__ __forceinline__ v8f v8f_zero() {
  v8f z = {0.f, 0.f, 0.f, 0.f, 0.f, 0.f, 0.f, 0.f};
  return z;
}

__device__ __forceinline__ u16 f32_bf16(float f) {
  union { float f; u32 u; } x; x.f = f;
  u32 u = x.u;
  u += 0x7FFFu + ((u >> 16) & 1u);   // round to nearest even
  return (u16)(u >> 16);
}

// ---------------------------------------------------------------------------
// CDNA5 async global->LDS staging (ASYNCcnt-tracked, bypasses VGPRs).
// Copies 32 contiguous bytes: INST_OFFSET is applied to BOTH the LDS dest
// and the global source (ISA §10, async pseudocode), so offset:16 advances
// both sides of the second b128 transfer.
// ---------------------------------------------------------------------------
__device__ __forceinline__ void async_copy32(u32 lds_byte, const void* g) {
  u64 ga = (u64)(uintptr_t)g;
  asm volatile(
      "global_load_async_to_lds_b128 %0, %1, off\n\t"
      "global_load_async_to_lds_b128 %0, %1, off offset:16"
      :
      : "v"(lds_byte), "v"(ga)
      : "memory");
}
__device__ __forceinline__ void async_wait0() {
  asm volatile("s_wait_asynccnt 0x0" ::: "memory");
}
__device__ __forceinline__ u32 lds_byte_addr(const void* p) {
  // Low 32 bits of a generic LDS pointer are the LDS byte offset.
  return (u32)(uintptr_t)p;
}

// ---------------------------------------------------------------------------
// RMSNorm (fp32 in) -> bf16 out, one row (D_=2048) per 256-thread block.
// ---------------------------------------------------------------------------
__global__ __launch_bounds__(256)
void rmsnorm_cast(const float* __restrict__ x, const float* __restrict__ w,
                  u16* __restrict__ out) {
  __shared__ float red[256];
  const int row = blockIdx.x, tid = threadIdx.x;
  const float* xr = x + (size_t)row * D_;
  float v[8];
  float ss = 0.f;
#pragma unroll
  for (int i = 0; i < 8; ++i) {
    v[i] = xr[tid + i * 256];
    ss += v[i] * v[i];
  }
  red[tid] = ss;
  __syncthreads();
  for (int s = 128; s > 0; s >>= 1) {
    if (tid < s) red[tid] += red[tid + s];
    __syncthreads();
  }
  const float scale = rsqrtf(red[0] * (1.0f / D_) + EPS_);
  u16* orow = out + (size_t)row * D_;
#pragma unroll
  for (int i = 0; i < 8; ++i)
    orow[tid + i * 256] = f32_bf16(v[i] * scale * w[tid + i * 256]);
}

// ---------------------------------------------------------------------------
// Transpose + cast: fp32 [R,C] row-major -> bf16 [C,R] row-major.
// ---------------------------------------------------------------------------
__global__ __launch_bounds__(256)
void transpose_cast(const float* __restrict__ in, u16* __restrict__ out,
                    int R, int C) {
  __shared__ float tile[32][33];
  const int c0 = blockIdx.x * 32, r0 = blockIdx.y * 32;
  const int tx = threadIdx.x, ty = threadIdx.y;   // 32 x 8
#pragma unroll
  for (int i = 0; i < 32; i += 8)
    tile[ty + i][tx] = in[(size_t)(r0 + ty + i) * C + (c0 + tx)];
  __syncthreads();
#pragma unroll
  for (int i = 0; i < 32; i += 8)
    out[(size_t)(c0 + ty + i) * R + (r0 + tx)] = f32_bf16(tile[tx][ty + i]);
}

// ---------------------------------------------------------------------------
// bf16 WMMA GEMM: C[M,N] = A[M,K] (bf16, row) * Bt[N,K] (bf16, row, i.e. B^T).
// 128x128 macro-tile, 8 waves (2x4), wave tile 64x32 (4x2 wmma accumulators).
// Staging uses GLOBAL_LOAD_ASYNC_TO_LDS_B128 (ASYNCcnt path).
// EPI 0: store bf16.  EPI 1: store fp32 = resid + acc (residual add).
// ---------------------------------------------------------------------------
template <int EPI>
__global__ __launch_bounds__(256)
void gemm_wmma(const u16* __restrict__ A, const u16* __restrict__ Bt,
               void* __restrict__ Cout, const float* __restrict__ resid,
               int M, int N, int K) {
  __shared__ u16 sA[128 * 40];
  __shared__ u16 sB[128 * 40];
  const int tid = threadIdx.x;
  const int lane = tid & 31, wv = tid >> 5;
  const int wm = (wv & 1) * 64, wn = (wv >> 1) * 32;
  const int mb = blockIdx.y * 128, nb = blockIdx.x * 128;
  const int ml = lane & 15, kh = lane >> 4;

  FragC acc[4][2];
#pragma unroll
  for (int i = 0; i < 4; ++i)
#pragma unroll
    for (int j = 0; j < 2; ++j) acc[i][j].v = v8f_zero();

  const int ld_r = tid >> 1;          // staging row 0..127
  const int ld_k = (tid & 1) * 16;    // 0 or 16
  const u16* gA = A + (size_t)(mb + ld_r) * K + ld_k;
  const u16* gB = Bt + (size_t)(nb + ld_r) * K + ld_k;
  const u32 ldsA = lds_byte_addr(&sA[ld_r * 40 + ld_k]);
  const u32 ldsB = lds_byte_addr(&sB[ld_r * 40 + ld_k]);

  for (int kb = 0; kb < K; kb += 32) {
    __syncthreads();
    async_copy32(ldsA, gA + kb);
    async_copy32(ldsB, gB + kb);
    __builtin_prefetch(gA + kb + 32, 0, 1);
    __builtin_prefetch(gB + kb + 32, 0, 1);
    async_wait0();
    __syncthreads();

    FragAB af[4], bfr[2];
#pragma unroll
    for (int i = 0; i < 4; ++i) {
      const u16* p = &sA[(wm + i * 16 + ml) * 40 + kh * 8];
      af[i].q[0] = *(const u32x4*)p;
      af[i].q[1] = *(const u32x4*)(p + 16);
    }
#pragma unroll
    for (int j = 0; j < 2; ++j) {
      const u16* p = &sB[(wn + j * 16 + ml) * 40 + kh * 8];
      bfr[j].q[0] = *(const u32x4*)p;
      bfr[j].q[1] = *(const u32x4*)(p + 16);
    }
#pragma unroll
    for (int i = 0; i < 4; ++i)
#pragma unroll
      for (int j = 0; j < 2; ++j)
        acc[i][j].v = __builtin_amdgcn_wmma_f32_16x16x32_bf16(
            false, af[i].v, false, bfr[j].v, (short)0, acc[i][j].v, false, false);
  }

#pragma unroll
  for (int i = 0; i < 4; ++i) {
#pragma unroll
    for (int r = 0; r < 8; ++r) {
      const int row = mb + wm + i * 16 + kh * 8 + r;
#pragma unroll
      for (int j = 0; j < 2; ++j) {
        const int col = nb + wn + j * 16 + ml;
        const size_t idx = (size_t)row * N + col;
        if (EPI == 0)
          ((u16*)Cout)[idx] = f32_bf16(acc[i][j].f[r]);
        else
          ((float*)Cout)[idx] = resid[idx] + acc[i][j].f[r];
      }
    }
  }
}

// ---------------------------------------------------------------------------
// Dual-B WMMA GEMM fused with SwiGLU epilogue:
//   gated = silu(A@B1^T + b1) * (A@B2^T + b2)   -> bf16
// ---------------------------------------------------------------------------
__global__ __launch_bounds__(256)
void gemm_wmma_swiglu(const u16* __restrict__ A, const u16* __restrict__ B1t,
                      const u16* __restrict__ B2t, const float* __restrict__ bias1,
                      const float* __restrict__ bias2, u16* __restrict__ Cout,
                      int M, int N, int K) {
  __shared__ u16 sA[128 * 40];
  __shared__ u16 sB1[128 * 40];
  __shared__ u16 sB2[128 * 40];
  const int tid = threadIdx.x;
  const int lane = tid & 31, wv = tid >> 5;
  const int wm = (wv & 1) * 64, wn = (wv >> 1) * 32;
  const int mb = blockIdx.y * 128, nb = blockIdx.x * 128;
  const int ml = lane & 15, kh = lane >> 4;

  FragC acc1[4][2], acc2[4][2];
#pragma unroll
  for (int i = 0; i < 4; ++i)
#pragma unroll
    for (int j = 0; j < 2; ++j) { acc1[i][j].v = v8f_zero(); acc2[i][j].v = v8f_zero(); }

  const int ld_r = tid >> 1;
  const int ld_k = (tid & 1) * 16;
  const u16* gA  = A   + (size_t)(mb + ld_r) * K + ld_k;
  const u16* gB1 = B1t + (size_t)(nb + ld_r) * K + ld_k;
  const u16* gB2 = B2t + (size_t)(nb + ld_r) * K + ld_k;
  const u32 ldsA  = lds_byte_addr(&sA[ld_r * 40 + ld_k]);
  const u32 ldsB1 = lds_byte_addr(&sB1[ld_r * 40 + ld_k]);
  const u32 ldsB2 = lds_byte_addr(&sB2[ld_r * 40 + ld_k]);

  for (int kb = 0; kb < K; kb += 32) {
    __syncthreads();
    async_copy32(ldsA, gA + kb);
    async_copy32(ldsB1, gB1 + kb);
    async_copy32(ldsB2, gB2 + kb);
    __builtin_prefetch(gA + kb + 32, 0, 1);
    async_wait0();
    __syncthreads();

    FragAB af[4], b1f[2], b2f[2];
#pragma unroll
    for (int i = 0; i < 4; ++i) {
      const u16* p = &sA[(wm + i * 16 + ml) * 40 + kh * 8];
      af[i].q[0] = *(const u32x4*)p;
      af[i].q[1] = *(const u32x4*)(p + 16);
    }
#pragma unroll
    for (int j = 0; j < 2; ++j) {
      const u16* p1 = &sB1[(wn + j * 16 + ml) * 40 + kh * 8];
      b1f[j].q[0] = *(const u32x4*)p1;
      b1f[j].q[1] = *(const u32x4*)(p1 + 16);
      const u16* p2 = &sB2[(wn + j * 16 + ml) * 40 + kh * 8];
      b2f[j].q[0] = *(const u32x4*)p2;
      b2f[j].q[1] = *(const u32x4*)(p2 + 16);
    }
#pragma unroll
    for (int i = 0; i < 4; ++i)
#pragma unroll
      for (int j = 0; j < 2; ++j) {
        acc1[i][j].v = __builtin_amdgcn_wmma_f32_16x16x32_bf16(
            false, af[i].v, false, b1f[j].v, (short)0, acc1[i][j].v, false, false);
        acc2[i][j].v = __builtin_amdgcn_wmma_f32_16x16x32_bf16(
            false, af[i].v, false, b2f[j].v, (short)0, acc2[i][j].v, false, false);
      }
  }

#pragma unroll
  for (int i = 0; i < 4; ++i) {
#pragma unroll
    for (int r = 0; r < 8; ++r) {
      const int row = mb + wm + i * 16 + kh * 8 + r;
#pragma unroll
      for (int j = 0; j < 2; ++j) {
        const int col = nb + wn + j * 16 + ml;
        const float h1 = acc1[i][j].f[r] + bias1[col];
        const float h2 = acc2[i][j].f[r] + bias2[col];
        const float g = (h1 / (1.f + __expf(-h1))) * h2;   // silu(h1)*h2
        ((u16*)Cout)[(size_t)row * N + col] = f32_bf16(g);
      }
    }
  }
}

// ---------------------------------------------------------------------------
// Flash attention (bf16 Q/K/V, fp32 online softmax, bf16 out).
// Grid: (S/128, H, B); block 256 (8 waves); wave owns 16 q rows.
// 32-key chunks: 8 WMMAs scores + 8 WMMAs P@V per chunk.
// ---------------------------------------------------------------------------
__global__ __launch_bounds__(256)
void flash_attn(const u16* __restrict__ Q, const u16* __restrict__ Kb,
                const u16* __restrict__ V, const float* __restrict__ mask,
                u16* __restrict__ O) {
  __shared__ u16 sVT[128 * 40];       // V^T chunk: [d][key], padded
  __shared__ u16 sP[8 * 16 * 40];     // per-wave P tile [16][40]
  const int b = blockIdx.z, h = blockIdx.y;
  const int qb = blockIdx.x * 128;
  const int tid = threadIdx.x, lane = tid & 31, wv = tid >> 5;
  const int ml = lane & 15, kh = lane >> 4;
  const float scl = 0.08838834764831845f;  // 1/sqrt(HD)

  // Q fragments (A layout), 4 chunks of K=32 across HD=128, kept in registers.
  FragAB qf[4];
  {
    const u16* qrow = Q + (size_t)(b * S_ + qb + wv * 16 + ml) * D_ + h * HD_;
#pragma unroll
    for (int c = 0; c < 4; ++c) {
      qf[c].q[0] = *(const u32x4*)(qrow + c * 32 + kh * 8);
      qf[c].q[1] = *(const u32x4*)(qrow + c * 32 + kh * 8 + 16);
    }
  }

  FragC oacc[8];
#pragma unroll
  for (int n = 0; n < 8; ++n) oacc[n].v = v8f_zero();
  float rowmax[8], rowsum[8];
#pragma unroll
  for (int r = 0; r < 8; ++r) { rowmax[r] = -3.0e38f; rowsum[r] = 0.f; }

  u16* myP = &sP[wv * 16 * 40];

  for (int kc = 0; kc < S_; kc += 32) {
    __syncthreads();
    // Cooperative stage of V^T chunk (32 keys x 128 d) into LDS (transposed,
    // so the explicit VGPR bounce path is required here).
    {
      const int key = tid & 31, dblk = tid >> 5;  // dblk 0..7
      const u16* vrow = V + (size_t)(b * S_ + kc + key) * D_ + h * HD_ + dblk * 16;
#pragma unroll
      for (int part = 0; part < 2; ++part) {
        U4H8 t; t.q = *(const u32x4*)(vrow + part * 8);
#pragma unroll
        for (int e = 0; e < 8; ++e)
          sVT[(dblk * 16 + part * 8 + e) * 40 + key] = t.h[e];
      }
    }
    __syncthreads();

    // Scores: two 16x16 tiles for this 32-key chunk (K fragments straight
    // from global; K is row-major so B-layout reads are contiguous b128s).
    FragC sc[2];
#pragma unroll
    for (int t2 = 0; t2 < 2; ++t2) {
      sc[t2].v = v8f_zero();
      const u16* krow =
          Kb + (size_t)(b * S_ + kc + t2 * 16 + ml) * D_ + h * HD_ + kh * 8;
#pragma unroll
      for (int c = 0; c < 4; ++c) {
        FragAB kf;
        kf.q[0] = *(const u32x4*)(krow + c * 32);
        kf.q[1] = *(const u32x4*)(krow + c * 32 + 16);
        sc[t2].v = __builtin_amdgcn_wmma_f32_16x16x32_bf16(
            false, qf[c].v, false, kf.v, (short)0, sc[t2].v, false, false);
      }
    }
    const float m0 = mask[b * S_ + kc + ml];
    const float m1 = mask[b * S_ + kc + 16 + ml];

    // Online softmax (row stats replicated across each 16-lane half).
#pragma unroll
    for (int r = 0; r < 8; ++r) {
      const float a0 = sc[0].f[r] * scl + m0;
      const float a1 = sc[1].f[r] * scl + m1;
      float mx = fmaxf(a0, a1);
#pragma unroll
      for (int off = 8; off >= 1; off >>= 1)
        mx = fmaxf(mx, __shfl_xor(mx, off, 32));
      const float nm = fmaxf(rowmax[r], mx);
      const float alpha = __expf(rowmax[r] - nm);
      rowmax[r] = nm;
      const float p0 = __expf(a0 - nm);
      const float p1 = __expf(a1 - nm);
      float ps = p0 + p1;
#pragma unroll
      for (int off = 8; off >= 1; off >>= 1)
        ps += __shfl_xor(ps, off, 32);
      rowsum[r] = rowsum[r] * alpha + ps;
      const int prow = kh * 8 + r;
      myP[prow * 40 + ml] = f32_bf16(p0);
      myP[prow * 40 + 16 + ml] = f32_bf16(p1);
#pragma unroll
      for (int n = 0; n < 8; ++n) oacc[n].f[r] *= alpha;
    }

    // Re-layout P from C layout to A layout via wave-private LDS bounce.
    FragAB pf;
    {
      const u16* p = &myP[ml * 40 + kh * 8];
      pf.q[0] = *(const u32x4*)p;
      pf.q[1] = *(const u32x4*)(p + 16);
    }
    // P@V over 8 HD chunks of 16.
#pragma unroll
    for (int n = 0; n < 8; ++n) {
      const u16* p = &sVT[(n * 16 + ml) * 40 + kh * 8];
      FragAB vf;
      vf.q[0] = *(const u32x4*)p;
      vf.q[1] = *(const u32x4*)(p + 16);
      oacc[n].v = __builtin_amdgcn_wmma_f32_16x16x32_bf16(
          false, pf.v, false, vf.v, (short)0, oacc[n].v, false, false);
    }
  }

  // Normalize and store bf16 attention output.
#pragma unroll
  for (int r = 0; r < 8; ++r) {
    const float inv = 1.0f / rowsum[r];
    const int row = qb + wv * 16 + kh * 8 + r;
    u16* orow = O + (size_t)(b * S_ + row) * D_ + h * HD_;
#pragma unroll
    for (int n = 0; n < 8; ++n)
      orow[n * 16 + ml] = f32_bf16(oacc[n].f[r] * inv);
  }
}

// ---------------------------------------------------------------------------
// Host-side launch sequence.
// ---------------------------------------------------------------------------
extern "C" void kernel_launch(void* const* d_in, const int* in_sizes, int n_in,
                              void* d_out, int out_size, void* d_ws, size_t ws_size,
                              hipStream_t stream) {
  (void)in_sizes; (void)n_in; (void)out_size; (void)ws_size;

  const float* hidden = (const float*)d_in[0];
  const float* mask   = (const float*)d_in[1];
  const float* rn1w   = (const float*)d_in[2];
  const float* wq     = (const float*)d_in[3];
  const float* wk     = (const float*)d_in[4];
  const float* wvp    = (const float*)d_in[5];
  const float* wo     = (const float*)d_in[6];
  const float* rn2w   = (const float*)d_in[7];
  const float* w1     = (const float*)d_in[8];
  const float* fc1w   = (const float*)d_in[9];
  const float* fc1b   = (const float*)d_in[10];
  const float* fc2w   = (const float*)d_in[11];
  const float* fc2b   = (const float*)d_in[12];
  const float* w3     = (const float*)d_in[13];

  char* ws = (char*)d_ws;
  size_t off = 0;
  auto alloc = [&](size_t bytes) -> char* {
    char* p = ws + off;
    off = (off + bytes + 255) & ~(size_t)255;
    return p;
  };

  const size_t SZ_TD   = (size_t)NT_ * D_ * 2;        // token x D bf16
  u16* xnorm  = (u16*)alloc(SZ_TD);                   // rmsnorm output (reused)
  u16* qbuf   = (u16*)alloc(SZ_TD);
  u16* kbuf   = (u16*)alloc(SZ_TD);
  u16* vbuf   = (u16*)alloc(SZ_TD);
  u16* attnO  = (u16*)alloc(SZ_TD);
  float* res1 = (float*)alloc((size_t)NT_ * D_ * 4);
  u16* wqT    = (u16*)alloc((size_t)D_ * D_ * 2);
  u16* wkT    = (u16*)alloc((size_t)D_ * D_ * 2);
  u16* wvT    = (u16*)alloc((size_t)D_ * D_ * 2);
  u16* woT    = (u16*)alloc((size_t)D_ * D_ * 2);
  u16* w1T    = (u16*)alloc((size_t)(2 * I_) * D_ * 2);
  u16* fc1T   = (u16*)alloc((size_t)I_ * (2 * I_) * 2);
  u16* fc2T   = (u16*)alloc((size_t)I_ * (2 * I_) * 2);
  u16* w3T    = (u16*)alloc((size_t)D_ * I_ * 2);
  // Dead-buffer reuse: xbuf (64MB) over q/k/v/attnO; gated (32MB) over wqT..woT.
  u16* xbuf   = qbuf;    // [NT_, 2I] bf16 = 64MB (q..attnO region)
  u16* gated  = wqT;     // [NT_, I]  bf16 = 32MB (wqT..woT region)

  const dim3 tb32x8(32, 8);

  // 1. attn_in = rmsnorm(hidden) -> bf16
  rmsnorm_cast<<<NT_, 256, 0, stream>>>(hidden, rn1w, xnorm);

  // 2. transpose-cast all weights to bf16 [N,K]
  transpose_cast<<<dim3(D_ / 32, D_ / 32), tb32x8, 0, stream>>>(wq, wqT, D_, D_);
  transpose_cast<<<dim3(D_ / 32, D_ / 32), tb32x8, 0, stream>>>(wk, wkT, D_, D_);
  transpose_cast<<<dim3(D_ / 32, D_ / 32), tb32x8, 0, stream>>>(wvp, wvT, D_, D_);
  transpose_cast<<<dim3(D_ / 32, D_ / 32), tb32x8, 0, stream>>>(wo, woT, D_, D_);
  transpose_cast<<<dim3((2 * I_) / 32, D_ / 32), tb32x8, 0, stream>>>(w1, w1T, D_, 2 * I_);
  transpose_cast<<<dim3(I_ / 32, (2 * I_) / 32), tb32x8, 0, stream>>>(fc1w, fc1T, 2 * I_, I_);
  transpose_cast<<<dim3(I_ / 32, (2 * I_) / 32), tb32x8, 0, stream>>>(fc2w, fc2T, 2 * I_, I_);
  transpose_cast<<<dim3(D_ / 32, I_ / 32), tb32x8, 0, stream>>>(w3, w3T, I_, D_);

  // 3. Q/K/V projections (bf16 out)
  gemm_wmma<0><<<dim3(D_ / 128, NT_ / 128), 256, 0, stream>>>(
      xnorm, wqT, (void*)qbuf, nullptr, NT_, D_, D_);
  gemm_wmma<0><<<dim3(D_ / 128, NT_ / 128), 256, 0, stream>>>(
      xnorm, wkT, (void*)kbuf, nullptr, NT_, D_, D_);
  gemm_wmma<0><<<dim3(D_ / 128, NT_ / 128), 256, 0, stream>>>(
      xnorm, wvT, (void*)vbuf, nullptr, NT_, D_, D_);

  // 4. flash attention
  flash_attn<<<dim3(S_ / 128, H_, B_), 256, 0, stream>>>(qbuf, kbuf, vbuf, mask, attnO);

  // 5. residual1 = hidden + attn @ wo   (fp32)
  gemm_wmma<1><<<dim3(D_ / 128, NT_ / 128), 256, 0, stream>>>(
      attnO, woT, (void*)res1, hidden, NT_, D_, D_);

  // 6. ffn_in = rmsnorm(residual1) -> bf16 (reuse xnorm)
  rmsnorm_cast<<<NT_, 256, 0, stream>>>(res1, rn2w, xnorm);

  // 7. x = ffn_in @ w1 -> bf16 [NT_, 2I]   (xbuf aliases dead q/k/v/attnO)
  gemm_wmma<0><<<dim3((2 * I_) / 128, NT_ / 128), 256, 0, stream>>>(
      xnorm, w1T, (void*)xbuf, nullptr, NT_, 2 * I_, D_);

  // 8. gated = silu(x@fc1 + b1) * (x@fc2 + b2) -> bf16 [NT_, I]
  gemm_wmma_swiglu<<<dim3(I_ / 128, NT_ / 128), 256, 0, stream>>>(
      xbuf, fc1T, fc2T, fc1b, fc2b, gated, NT_, I_, 2 * I_);

  // 9. out = residual1 + gated @ w3   (fp32 -> d_out)
  gemm_wmma<1><<<dim3(D_ / 128, NT_ / 128), 256, 0, stream>>>(
      gated, w3T, d_out, res1, NT_, D_, I_);
}
